// EdgeFeatureGATConv_33861522162216
// MI455X (gfx1250) — matrix operand order repaired and checked by reading, not compiled, not run on hardware
//
#include <hip/hip_runtime.h>
#include <hip/hip_bf16.h>

// ---------------------------------------------------------------------------
// EdgeFeatureGATConv for MI455X (gfx1250, wave32, WMMA)
//
//   ea  = MLP(edge_attr)                       (3->6->3, per edge)
//   mean_ea[d] = segment_mean(ea, dst)         (self-loop fill)
//   xh  = x @ W_lin                            (WMMA f16 in / f32 acc)
//   a_src/a_dst = <xh, att_src/att_dst> per head
//   a_edge = ea @ v,   v[j][h] = sum_c W_edge[j,h*32+c]*att_edge[h,c]
//   alpha = leaky_relu(a_src[s]+a_dst[d]+a_edge, 0.2)
//   segment softmax over dst, msg scatter-add, + bias
// Outputs concatenated: out [N*128] then att [(E+N)*4]
// ---------------------------------------------------------------------------

typedef __attribute__((ext_vector_type(16))) _Float16 v16h;
typedef __attribute__((ext_vector_type(8)))  _Float16 v8h;
typedef __attribute__((ext_vector_type(8)))  float    v8f;

#define HC    128   // H*C
#define INCH  128   // in_channels
#define HEADS 4
#define CPH   32
#define SLOPE 0.2f
#define GROWS 64    // rows per GEMM block (4 row-tiles of 16)

// ---- monotone encoding of float for atomicMax on unsigned ------------------
__device__ __forceinline__ unsigned fenc(float f) {
  unsigned u = __float_as_uint(f);
  return (u & 0x80000000u) ? ~u : (u | 0x80000000u);
}
__device__ __forceinline__ float fdec(unsigned e) {
  unsigned u = (e & 0x80000000u) ? (e & 0x7FFFFFFFu) : ~e;
  return __uint_as_float(u);
}

// ---------------------------------------------------------------------------
// K0: init. out <- bias broadcast; zero deg/sum_ea/denom; amax <- 0 (== -inf
// in the monotone-uint encoding since every encoded real value is > 0).
// ---------------------------------------------------------------------------
__global__ void k_init(float* __restrict__ out, const float* __restrict__ bias,
                       float* __restrict__ deg, float* __restrict__ sum_ea,
                       float* __restrict__ denom, unsigned* __restrict__ amax,
                       int Nn) {
  int idx = blockIdx.x * blockDim.x + threadIdx.x;
  long long tot = (long long)Nn * HC;
  if (idx < tot) out[idx] = bias[idx & (HC - 1)];
  if (idx < Nn) deg[idx] = 0.0f;
  if (idx < Nn * 3) sum_ea[idx] = 0.0f;
  if (idx < Nn * HEADS) { denom[idx] = 0.0f; amax[idx] = 0u; }
}

// ---------------------------------------------------------------------------
// K1: edge MLP (3 -> 6 -> 3) + per-destination degree / ea sums (atomics).
// ---------------------------------------------------------------------------
__global__ void k_edge_mlp(const float* __restrict__ edge_attr,
                           const int* __restrict__ dst,
                           const float* __restrict__ We1, const float* __restrict__ be1,
                           const float* __restrict__ We2, const float* __restrict__ be2,
                           float* __restrict__ ea, float* __restrict__ deg,
                           float* __restrict__ sum_ea, int E) {
  int e = blockIdx.x * blockDim.x + threadIdx.x;
  if (e >= E) return;
  float i0 = edge_attr[e * 3 + 0];
  float i1 = edge_attr[e * 3 + 1];
  float i2 = edge_attr[e * 3 + 2];
  float h[6];
#pragma unroll
  for (int j = 0; j < 6; ++j) {
    float t = i0 * We1[0 * 6 + j] + i1 * We1[1 * 6 + j] + i2 * We1[2 * 6 + j] + be1[j];
    h[j] = t > 0.0f ? t : 0.0f;
  }
  int d = dst[e];
  atomicAdd(&deg[d], 1.0f);
#pragma unroll
  for (int k = 0; k < 3; ++k) {
    float o = be2[k];
#pragma unroll
    for (int j = 0; j < 6; ++j) o += h[j] * We2[j * 3 + k];
    ea[e * 3 + k] = o;
    atomicAdd(&sum_ea[d * 3 + k], o);
  }
}

// ---------------------------------------------------------------------------
// K2: v[j][h] = sum_c W_edge[j, h*32+c] * att_edge[h, c]   (12 values)
// ---------------------------------------------------------------------------
__global__ void k_edge_vec(const float* __restrict__ W_edge,
                           const float* __restrict__ att_edge,
                           float* __restrict__ v) {
  int idx = threadIdx.x;
  if (idx >= 12) return;
  int j = idx >> 2, hh = idx & 3;
  float s = 0.0f;
  for (int c = 0; c < CPH; ++c)
    s += W_edge[j * HC + hh * CPH + c] * att_edge[hh * CPH + c];
  v[idx] = s;
}

// ---------------------------------------------------------------------------
// K2b: Wt[n][k] = (f16)W_lin[k][n]  — one-shot transpose + f16 convert so the
// GEMM's B fragments are 32B-contiguous per lane (2x global_load_b128, L2-hit).
// ---------------------------------------------------------------------------
__global__ void k_prep_wt(const float* __restrict__ W, _Float16* __restrict__ Wt) {
  int idx = blockIdx.x * blockDim.x + threadIdx.x;
  if (idx >= INCH * HC) return;
  int k = idx >> 7, n = idx & (HC - 1);
  Wt[(size_t)n * INCH + k] = (_Float16)W[(size_t)k * HC + n];
}

// ---------------------------------------------------------------------------
// K3: WMMA GEMM  xh[N,128] = x[N,128] @ W_lin[128,128]
// Block: 256 threads = 8 waves, covers GROWS(64) rows x 128 cols.
// Wave w owns columns [16w,16w+16) and sweeps 4 row-tiles, holding its four
// B fragments (K swept in 4 steps of 32) in VGPRs for the whole block.
// A panel (64x128 f16 = 16KB) staged in LDS with f32->f16 conversion;
// A fragments are 16B-aligned v8h vector loads (ds_load_b128).
// Row bounds are handled by a single BLOCK-UNIFORM branch (only the last
// block can be partial) so the hot path has no per-element exec masking.
//
// VGPR layouts (wave32; A/B in this code's contiguous k order), m=lane%16, hl=lane/16:
//   A (16x32 f16): a[i]   = A[m][K0 + hl*16 + i]     i=0..7
//                  a[8+i] = A[m][K0 + hl*16 + 8 + i]
//   B (32x16 f16): b[i]   = B[K0 + hl*16 + i][m]     i=0..15  (contig in Wt)
//   C/D (16x16 f32): c[r] = D[hl*8 + r][m]
// ---------------------------------------------------------------------------
__global__ void k_gemm_wmma(const float* __restrict__ x,
                            const _Float16* __restrict__ Wt,
                            float* __restrict__ xh, int Nn) {
  __shared__ _Float16 sA[GROWS * INCH];   // 64x128 f16 = 16 KB

  const int tid  = threadIdx.x;
  const int row0 = blockIdx.x * GROWS;
  const bool full = (row0 + GROWS) <= Nn;   // block-uniform

  // stage this block's 64-row A panel (f32 -> f16), 32 elements per thread
  if (full) {
    const float* xp = x + (size_t)row0 * INCH;
    for (int t = tid; t < GROWS * INCH; t += 256) sA[t] = (_Float16)xp[t];
  } else {
    for (int t = tid; t < GROWS * INCH; t += 256) {
      int gr = row0 + (t >> 7);
      sA[t] = (_Float16)(gr < Nn ? x[(size_t)gr * INCH + (t & (INCH - 1))] : 0.0f);
    }
  }
  __syncthreads();

  const int lane = tid & 31;
  const int m    = lane & 15;
  const int hl   = lane >> 4;
  const int col0 = (tid >> 5) * 16;

  // hoist B: 4 contiguous 32B fragments per lane from the transposed f16 W
  v16h b[4];
  const _Float16* wrow = Wt + (size_t)(col0 + m) * INCH + hl * 16;
#pragma unroll
  for (int kk = 0; kk < 4; ++kk)
    b[kk] = *(const v16h*)(wrow + kk * 32);

#pragma unroll
  for (int rt = 0; rt < GROWS / 16; ++rt) {
    const _Float16* arow = sA + (size_t)(rt * 16 + m) * INCH + hl * 16;
    v8f c = {};
#pragma unroll
    for (int kk = 0; kk < 4; ++kk) {
      v8h alo = *(const v8h*)(arow + kk * 32);
      v8h ahi = *(const v8h*)(arow + kk * 32 + 8);
      v16h a = __builtin_shufflevector(alo, ahi,
                                       0, 1, 2, 3, 4, 5, 6, 7,
                                       8, 9, 10, 11, 12, 13, 14, 15);
      c = __builtin_amdgcn_wmma_f32_16x16x32_f16(
          /*neg_a=*/false, a, /*neg_b=*/false, b[kk],
          /*c_mod=*/(short)0, c, /*reuse_a=*/false, /*reuse_b=*/false);
    }
    const int orow = row0 + rt * 16 + hl * 8;
    float* op = xh + (size_t)orow * HC + col0 + m;
    if (full) {
#pragma unroll
      for (int r = 0; r < 8; ++r) op[(size_t)r * HC] = c[r];
    } else {
#pragma unroll
      for (int r = 0; r < 8; ++r)
        if (orow + r < Nn) op[(size_t)r * HC] = c[r];
    }
  }
}

// ---------------------------------------------------------------------------
// K4: per-node attention logits a_src/a_dst  (one thread per (n,h))
// ---------------------------------------------------------------------------
__global__ void k_node_att(const float* __restrict__ xh,
                           const float* __restrict__ att_src,
                           const float* __restrict__ att_dst,
                           float* __restrict__ a_src, float* __restrict__ a_dst,
                           int Nn) {
  int idx = blockIdx.x * blockDim.x + threadIdx.x;
  if (idx >= Nn * HEADS) return;
  int n = idx >> 2, hh = idx & 3;
  const float* xr = xh + (size_t)n * HC + hh * CPH;
  const float* as = att_src + hh * CPH;
  const float* ad = att_dst + hh * CPH;
  float s = 0.0f, d = 0.0f;
#pragma unroll
  for (int cc = 0; cc < CPH; ++cc) { float vv = xr[cc]; s += vv * as[cc]; d += vv * ad[cc]; }
  a_src[idx] = s;
  a_dst[idx] = d;
}

// ---------------------------------------------------------------------------
// K5: alpha = leaky_relu(a_src[s]+a_dst[d]+ea·v), segment max via atomicMax.
// i in [0, E+N); i >= E are self loops using mean edge attr.
// ---------------------------------------------------------------------------
__global__ void k_alpha(const int* __restrict__ src, const int* __restrict__ dst,
                        const float* __restrict__ ea, const float* __restrict__ sum_ea,
                        const float* __restrict__ deg, const float* __restrict__ a_src,
                        const float* __restrict__ a_dst, const float* __restrict__ v,
                        float* __restrict__ alpha, unsigned* __restrict__ amax,
                        int E, int Nn) {
  int i = blockIdx.x * blockDim.x + threadIdx.x;
  if (i >= E + Nn) return;
  int s, d; float e0, e1, e2;
  if (i < E) {
    s = src[i]; d = dst[i];
    e0 = ea[i * 3 + 0]; e1 = ea[i * 3 + 1]; e2 = ea[i * 3 + 2];
  } else {
    int n = i - E; s = n; d = n;
    float dg = fmaxf(deg[n], 1.0f);
    e0 = sum_ea[n * 3 + 0] / dg; e1 = sum_ea[n * 3 + 1] / dg; e2 = sum_ea[n * 3 + 2] / dg;
  }
#pragma unroll
  for (int hh = 0; hh < HEADS; ++hh) {
    float aeg = e0 * v[0 * 4 + hh] + e1 * v[1 * 4 + hh] + e2 * v[2 * 4 + hh];
    float al = a_src[s * HEADS + hh] + a_dst[d * HEADS + hh] + aeg;
    al = al > 0.0f ? al : SLOPE * al;
    alpha[(size_t)i * HEADS + hh] = al;
    atomicMax(&amax[d * HEADS + hh], fenc(al));
  }
}

// ---------------------------------------------------------------------------
// K6: ex = exp(alpha - amax[d]); overwrite alpha buffer; atomic denom sum.
// ---------------------------------------------------------------------------
__global__ void k_expsum(const int* __restrict__ dst, float* __restrict__ alpha,
                         const unsigned* __restrict__ amax,
                         float* __restrict__ denom, int E, int Nn) {
  int i = blockIdx.x * blockDim.x + threadIdx.x;
  if (i >= E + Nn) return;
  int d = (i < E) ? dst[i] : (i - E);
#pragma unroll
  for (int hh = 0; hh < HEADS; ++hh) {
    float am = fdec(amax[d * HEADS + hh]);
    float ex = __expf(alpha[(size_t)i * HEADS + hh] - am);
    alpha[(size_t)i * HEADS + hh] = ex;
    atomicAdd(&denom[d * HEADS + hh], ex);
  }
}

// ---------------------------------------------------------------------------
// K7: one wave per edge: att = ex/denom[d]; write att output; gather xh[src]
// as float4 per lane (128 floats / 32 lanes) and atomic-add into out[dst].
// ---------------------------------------------------------------------------
__global__ void k_scatter(const int* __restrict__ src, const int* __restrict__ dst,
                          const float* __restrict__ xh, const float* __restrict__ ex,
                          const float* __restrict__ denom, float* __restrict__ out,
                          float* __restrict__ att_out, int E, int Nn) {
  long long gid = (long long)blockIdx.x * blockDim.x + threadIdx.x;
  int i = (int)(gid >> 5);
  int lane = (int)(gid & 31);
  if (i >= E + Nn) return;
  int s, d;
  if (i < E) { s = src[i]; d = dst[i]; } else { s = i - E; d = s; }

  int c4 = lane * 4;
  int hh = lane >> 3;                       // 8 lanes per head (32 ch / 4)
  float att = ex[(size_t)i * HEADS + hh] / denom[d * HEADS + hh];
  if (lane < HEADS)
    att_out[(size_t)i * HEADS + lane] =
        ex[(size_t)i * HEADS + lane] / denom[d * HEADS + lane];

  const float4 xv = *(const float4*)(xh + (size_t)s * HC + c4);
  float* op = out + (size_t)d * HC + c4;
  atomicAdd(op + 0, xv.x * att);
  atomicAdd(op + 1, xv.y * att);
  atomicAdd(op + 2, xv.z * att);
  atomicAdd(op + 3, xv.w * att);
}

// ---------------------------------------------------------------------------
extern "C" void kernel_launch(void* const* d_in, const int* in_sizes, int n_in,
                              void* d_out, int out_size, void* d_ws, size_t ws_size,
                              hipStream_t stream) {
  const float* x         = (const float*)d_in[0];
  const int*   edge_index= (const int*)  d_in[1];
  const float* edge_attr = (const float*)d_in[2];
  const float* We1       = (const float*)d_in[3];
  const float* be1       = (const float*)d_in[4];
  const float* We2       = (const float*)d_in[5];
  const float* be2       = (const float*)d_in[6];
  const float* W_lin     = (const float*)d_in[7];
  const float* att_src   = (const float*)d_in[8];
  const float* att_dst   = (const float*)d_in[9];
  const float* W_edge    = (const float*)d_in[10];
  const float* att_edge  = (const float*)d_in[11];
  const float* bias      = (const float*)d_in[12];

  const int Nn = in_sizes[0] / INCH;   // 100000
  const int E  = in_sizes[1] / 2;      // 1600000
  const int* src = edge_index;
  const int* dst = edge_index + E;

  // ---- workspace carve-up (floats) ----
  float* ws = (float*)d_ws;
  size_t o = 0;
  float*    xh     = ws + o; o += (size_t)Nn * HC;          // 12.8M
  float*    ea     = ws + o; o += (size_t)E * 3;            //  4.8M
  float*    deg    = ws + o; o += (size_t)Nn;               //  0.1M
  float*    sum_ea = ws + o; o += (size_t)Nn * 3;           //  0.3M
  float*    a_src  = ws + o; o += (size_t)Nn * HEADS;       //  0.4M
  float*    a_dst  = ws + o; o += (size_t)Nn * HEADS;       //  0.4M
  float*    vvec   = ws + o; o += 16;
  float*    alpha  = ws + o; o += (size_t)(E + Nn) * HEADS; //  6.8M (alpha->ex)
  unsigned* amax   = (unsigned*)(ws + o); o += (size_t)Nn * HEADS;
  float*    denom  = ws + o; o += (size_t)Nn * HEADS;
  _Float16* Wt     = (_Float16*)(ws + o); o += (INCH * HC) / 2; // f16 transposed W
  (void)ws_size; (void)n_in; (void)out_size;

  float* out_nodes = (float*)d_out;                   // [N*128]
  float* out_att   = (float*)d_out + (size_t)Nn * HC; // [(E+N)*4]

  const int T = 256;

  // K0: init out<-bias, zero accumulators
  {
    long long tot = (long long)Nn * HC;
    int blocks = (int)((tot + T - 1) / T);
    k_init<<<blocks, T, 0, stream>>>(out_nodes, bias, deg, sum_ea, denom, amax, Nn);
  }
  // K1: edge MLP + degree/sum
  k_edge_mlp<<<(E + T - 1) / T, T, 0, stream>>>(edge_attr, dst, We1, be1, We2, be2,
                                                ea, deg, sum_ea, E);
  // K2: collapsed edge-attention vector v[3][4]
  k_edge_vec<<<1, 32, 0, stream>>>(W_edge, att_edge, vvec);
  // K2b: transposed f16 W for contiguous B fragments
  k_prep_wt<<<(INCH * HC + T - 1) / T, T, 0, stream>>>(W_lin, Wt);
  // K3: WMMA GEMM xh = x @ W_lin
  k_gemm_wmma<<<(Nn + GROWS - 1) / GROWS, T, 0, stream>>>(x, Wt, xh, Nn);
  // K4: per-node a_src / a_dst
  k_node_att<<<(Nn * HEADS + T - 1) / T, T, 0, stream>>>(xh, att_src, att_dst,
                                                         a_src, a_dst, Nn);
  // K5: alpha + segment max
  k_alpha<<<(E + Nn + T - 1) / T, T, 0, stream>>>(src, dst, ea, sum_ea, deg,
                                                  a_src, a_dst, vvec, alpha, amax, E, Nn);
  // K6: exp + denom
  k_expsum<<<(E + Nn + T - 1) / T, T, 0, stream>>>(dst, alpha, amax, denom, E, Nn);
  // K7: normalize, write att, scatter messages (wave per edge)
  {
    long long thr = (long long)(E + Nn) * 32;
    int blocks = (int)((thr + T - 1) / T);
    k_scatter<<<blocks, T, 0, stream>>>(src, dst, xh, alpha, denom,
                                        out_nodes, out_att, E, Nn);
  }
}
